// NTMCell_61091614818691
// MI455X (gfx1250) — matrix operand order repaired and checked by reading, not compiled, not run on hardware
//
#include <hip/hip_runtime.h>
#include <hip/hip_bf16.h>

// ---------------------------------------------------------------------------
// Problem constants (from reference)
// ---------------------------------------------------------------------------
#define BATCH   2048
#define NMEM    256
#define MMEM    64
#define UDIM    512
#define INDIM   10
#define KBIG    (INDIM + UDIM + UDIM)   // 1034
#define KBIGP   1056                    // KBIG padded to multiple of 32
#define ZCOLS   (4 * UDIM)              // 2048
#define HEADW   268                     // (M+6) + (3M+6)
#define HEADP   320                     // padded to multiple of 64
#define XOCOLS  (UDIM + MMEM)           // 576
#define CLIPV   20.0f

typedef __attribute__((ext_vector_type(16))) __bf16 v16bf;
typedef __attribute__((ext_vector_type(8)))  float  v8f;

// ---------------------------------------------------------------------------
// Helpers
// ---------------------------------------------------------------------------
static __device__ __forceinline__ unsigned short f32_to_bf16(float f) {
    unsigned int u = __float_as_uint(f);
    unsigned int r = (u + 0x7FFFu + ((u >> 16) & 1u)) >> 16;   // RNE
    return (unsigned short)r;
}
static __device__ __forceinline__ float sigmoidf_(float x) {
    return 1.0f / (1.0f + __expf(-x));
}
static __device__ __forceinline__ float softplusf_(float x) {
    return fmaxf(x, 0.0f) + log1pf(__expf(-fabsf(x)));
}

// ---------------------------------------------------------------------------
// f32 -> bf16 strided converter (builds concatenated bf16 operands)
// ---------------------------------------------------------------------------
__global__ void cvt_f32_bf16_kernel(const float* __restrict__ src, int sld,
                                    unsigned short* __restrict__ dst, int dld,
                                    int rows, int cols) {
    long idx = (long)blockIdx.x * 256 + threadIdx.x;
    long total = (long)rows * cols;
    if (idx >= total) return;
    int r = (int)(idx / cols), c = (int)(idx % cols);
    dst[(size_t)r * dld + c] = f32_to_bf16(src[(size_t)r * sld + c]);
}

// f32 [srows, scols] -> bf16 transposed [scols, dld>=srows] (dst[c][r] = src[r][c])
__global__ void cvt_t_f32_bf16_kernel(const float* __restrict__ src, int sld,
                                      int srows, int scols,
                                      unsigned short* __restrict__ dst, int dld) {
    long idx = (long)blockIdx.x * 256 + threadIdx.x;
    long total = (long)srows * scols;
    if (idx >= total) return;
    int c = (int)(idx / srows);   // dst row (N index)
    int r = (int)(idx % srows);   // dst col (K index) -> coalesced dst writes
    dst[(size_t)c * dld + r] = f32_to_bf16(src[(size_t)r * sld + c]);
}

__global__ void pack_bias_kernel(const float* __restrict__ br,
                                 const float* __restrict__ bw,
                                 float* __restrict__ dst) {
    int t = blockIdx.x * 256 + threadIdx.x;
    if (t < MMEM + 6) dst[t] = br[t];
    else if (t < HEADW) dst[t] = bw[t - (MMEM + 6)];
}

// ---------------------------------------------------------------------------
// bf16 WMMA GEMM:  C[M,N] = A[M,K] @ B[K,N] + bias   (B supplied TRANSPOSED)
//   Block = 256 threads (8 wave32), C tile 128x64, K-step 32.
//   Each wave computes a 32x32 micro-tile -> 4 v_wmma_f32_16x16x32_bf16.
//   All dims pre-padded by the caller: no bounds checks anywhere.
//   LDS tiles are pre-shuffled into the wave32 fragment layout (ISA 7.12.2);
//   consecutive even-aligned k pairs share a lane and consecutive 'e', so
//   staging uses plain b32 global loads + b32 LDS stores.
// ---------------------------------------------------------------------------
#define GEMM_OUT_BF16 1
#define GEMM_CLIP     2

__global__ __launch_bounds__(256)
void gemm_bf16_wmma_kernel(const unsigned short* __restrict__ A,  int lda,
                           const unsigned short* __restrict__ Bt, int ldbt,
                           const float* __restrict__ bias,
                           void* __restrict__ Cout, int ldc,
                           int K, int flags) {
    __shared__ __align__(32) unsigned short lA[8][32][16];  // 8 KB (128x32)
    __shared__ __align__(32) unsigned short lB[4][32][16];  // 4 KB (32x64)

    const int t    = threadIdx.x;
    const int wid  = t >> 5;
    const int lane = t & 31;
    const int wm   = wid >> 1;          // 0..3 : 32-row strip
    const int wn   = wid & 1;           // 0..1 : 32-col strip
    const int row0 = blockIdx.y * 128;
    const int col0 = blockIdx.x * 64;

    v8f acc[2][2] = {};

    for (int k0 = 0; k0 < K; k0 += 32) {
        // ---- stage A tile (128x32): 8 x b32 per thread, fragment-shuffled ----
        #pragma unroll
        for (int i = 0; i < 8; ++i) {
            int U = t + i * 256;              // uint index, 0..2047
            int m = U >> 4, k = (U & 15) * 2;
            unsigned int v = *(const unsigned int*)
                (A + (size_t)(row0 + m) * lda + k0 + k);
            int lane_s = (m & 15) + (((k >> 3) & 1) << 4);
            int e      = (k & 7) + ((k >= 16) ? 8 : 0);
            *(unsigned int*)(&lA[m >> 4][lane_s][e]) = v;
        }
        // ---- stage B tile (64 n x 32 k from B^T): 4 x b32 per thread ----
        #pragma unroll
        for (int i = 0; i < 4; ++i) {
            int U = t + i * 256;              // uint index, 0..1023
            int n = U >> 4, k = (U & 15) * 2;
            unsigned int v = *(const unsigned int*)
                (Bt + (size_t)(col0 + n) * ldbt + k0 + k);
            int lane_s = (n & 15) + (((k >> 3) & 1) << 4);
            int e      = (k & 7) + ((k >= 16) ? 8 : 0);
            *(unsigned int*)(&lB[n >> 4][lane_s][e]) = v;
        }
        __syncthreads();

        v16bf af[2], bfr[2];
        af[0]  = *reinterpret_cast<const v16bf*>(&lA[wm * 2][lane][0]);
        af[1]  = *reinterpret_cast<const v16bf*>(&lA[wm * 2 + 1][lane][0]);
        bfr[0] = *reinterpret_cast<const v16bf*>(&lB[wn * 2][lane][0]);
        bfr[1] = *reinterpret_cast<const v16bf*>(&lB[wn * 2 + 1][lane][0]);
        #pragma unroll
        for (int am = 0; am < 2; ++am)
            #pragma unroll
            for (int bn = 0; bn < 2; ++bn)
                acc[am][bn] = __builtin_amdgcn_wmma_f32_16x16x32_bf16(
                    false, af[am], false, bfr[bn], (short)0, acc[am][bn],
                    false, false);
        __syncthreads();
    }

    // ---- epilogue: C/D layout: m = r + 8*(lane>=16), n = lane%16 ----
    const int lhi = (lane >= 16) ? 8 : 0;
    #pragma unroll
    for (int am = 0; am < 2; ++am) {
        int mb = row0 + wm * 32 + am * 16 + lhi;
        #pragma unroll
        for (int bn = 0; bn < 2; ++bn) {
            int gn = col0 + wn * 32 + bn * 16 + (lane & 15);
            float bv = bias[gn];
            #pragma unroll
            for (int r = 0; r < 8; ++r) {
                float v = acc[am][bn][r] + bv;
                if (flags & GEMM_CLIP) v = fminf(fmaxf(v, -CLIPV), CLIPV);
                if (flags & GEMM_OUT_BF16)
                    ((unsigned short*)Cout)[(size_t)(mb + r) * ldc + gn] = f32_to_bf16(v);
                else
                    ((float*)Cout)[(size_t)(mb + r) * ldc + gn] = v;
            }
        }
    }
}

// ---------------------------------------------------------------------------
// Fused LSTM gates: z[B,4U], C0[B,U] -> h (clipped) as bf16 into Xo[:,0:U]
// ---------------------------------------------------------------------------
__global__ void lstm_kernel(const float* __restrict__ z,
                            const float* __restrict__ C0,
                            unsigned short* __restrict__ Xo) {
    int idx = blockIdx.x * 256 + threadIdx.x;
    if (idx >= BATCH * UDIM) return;
    int b = idx >> 9, u = idx & (UDIM - 1);
    const float* zb = z + (size_t)b * ZCOLS;
    float iv = sigmoidf_(zb[u]);
    float fv = sigmoidf_(zb[UDIM + u]);
    float gv = tanhf(zb[2 * UDIM + u]);
    float ov = sigmoidf_(zb[3 * UDIM + u]);
    float c  = fv * C0[idx] + iv * gv;
    float h  = ov * tanhf(c);
    h = fminf(fmaxf(h, -CLIPV), CLIPV);
    Xo[(size_t)b * XOCOLS + u] = f32_to_bf16(h);
}

// ---------------------------------------------------------------------------
// Block-wide reductions over 256 threads
// ---------------------------------------------------------------------------
static __device__ __forceinline__ float block_max(float* red, int t, float v) {
    red[t] = v; __syncthreads();
    for (int s = 128; s > 0; s >>= 1) {
        if (t < s) red[t] = fmaxf(red[t], red[t + s]);
        __syncthreads();
    }
    float r = red[0]; __syncthreads();
    return r;
}
static __device__ __forceinline__ float block_sum(float* red, int t, float v) {
    red[t] = v; __syncthreads();
    for (int s = 128; s > 0; s >>= 1) {
        if (t < s) red[t] += red[t + s];
        __syncthreads();
    }
    float r = red[0]; __syncthreads();
    return r;
}

// ---------------------------------------------------------------------------
// NTM addressing + memory update. One block per batch element; m0[b] (64 KB)
// staged into LDS once -> HBM sees m0 read once + m_t written once (the floor).
// ---------------------------------------------------------------------------
__global__ __launch_bounds__(256)
void addressing_kernel(const float* __restrict__ heads,  // [B, HEADP]
                       const float* __restrict__ A0,     // [2, B, N]
                       const float* __restrict__ m0,     // [B, N, M]
                       float* __restrict__ out_mt,       // [B, N, M]
                       float* __restrict__ out_Rt,       // [B, M]
                       float* __restrict__ out_wr,       // [B, N]
                       float* __restrict__ out_ww,       // [B, N]
                       unsigned short* __restrict__ XoR) // Xo+512, ld 576
{
    __shared__ float lm[NMEM * MMEM];   // 64 KB m0 slice
    __shared__ float kvec[MMEM];
    __shared__ float wg[NMEM];
    __shared__ float red[NMEM];
    __shared__ float wsel[2][NMEM];
    __shared__ float delv[MMEM], addv[MMEM];

    const int b = blockIdx.x;
    const int t = threadIdx.x;
    const float* mb = m0 + (size_t)b * (NMEM * MMEM);
    const float* hb = heads + (size_t)b * HEADP;

    #pragma unroll 4
    for (int i = 0; i < 64; ++i) lm[i * 256 + t] = mb[i * 256 + t];
    if (t < MMEM) {
        delv[t] = sigmoidf_(hb[140 + t]);   // w_out[:, 70:134]
        addv[t] = tanhf(hb[204 + t]);       // w_out[:, 134:198]
    }
    __syncthreads();

    const float* row = &lm[t * MMEM];
    float nn = 0.0f;
    #pragma unroll 8
    for (int m = 0; m < MMEM; ++m) { float v = row[m]; nn += v * v; }

    for (int head = 0; head < 2; ++head) {
        const int off = head * (MMEM + 6);
        if (t < MMEM) kvec[t] = tanhf(hb[off + t]);
        __syncthreads();

        float beta  = softplusf_(hb[off + MMEM]);
        float g     = sigmoidf_(hb[off + MMEM + 1]);
        float s0 = hb[off + MMEM + 2], s1 = hb[off + MMEM + 3], s2 = hb[off + MMEM + 4];
        float gamma = softplusf_(hb[off + MMEM + 5]);
        float smax = fmaxf(s0, fmaxf(s1, s2));
        float e0 = __expf(s0 - smax), e1 = __expf(s1 - smax), e2 = __expf(s2 - smax);
        float esum = e0 + e1 + e2;
        s0 = e0 / esum; s1 = e1 / esum; s2 = e2 / esum;

        float ip = 0.0f, k2 = 0.0f;
        #pragma unroll 8
        for (int m = 0; m < MMEM; ++m) { float kv = kvec[m]; ip += row[m] * kv; k2 += kv * kv; }
        float Kn = ip / (sqrtf(k2) * sqrtf(nn) + 1e-8f);

        float xv = beta * Kn;
        float mx = block_max(red, t, xv);
        float ex = __expf(xv - mx);
        float sm = block_sum(red, t, ex);
        float wc = ex / sm;

        float ap  = A0[((size_t)head * BATCH + b) * NMEM + t];
        float wgv = g * wc + (1.0f - g) * ap;
        wg[t] = wgv; __syncthreads();
        float conv = s0 * wgv + s1 * wg[(t + NMEM - 1) & (NMEM - 1)]
                              + s2 * wg[(t + 1) & (NMEM - 1)];
        conv = fmaxf(conv, 0.0f);
        float wsh = powf(conv, gamma);
        float wsum = block_sum(red, t, wsh);
        float wfin = wsh / wsum;
        wsel[head][t] = wfin;
        (head == 0 ? out_wr : out_ww)[(size_t)b * NMEM + t] = wfin;
        __syncthreads();
    }

    if (t < MMEM) {
        float acc = 0.0f;
        for (int n = 0; n < NMEM; ++n) acc += wsel[0][n] * lm[n * MMEM + t];
        out_Rt[(size_t)b * MMEM + t] = acc;
        XoR[(size_t)b * XOCOLS + t]  = f32_to_bf16(acc);
    }
    __syncthreads();

    float* mtb = out_mt + (size_t)b * (NMEM * MMEM);
    #pragma unroll 4
    for (int i = 0; i < 64; ++i) {
        int idx = i * 256 + t;
        int n = idx >> 6, m = idx & (MMEM - 1);
        float ww = wsel[1][n];
        mtb[idx] = lm[idx] * (1.0f - ww * delv[m]) + ww * addv[m];
    }
}

// ---------------------------------------------------------------------------
// Host-side launcher
// ---------------------------------------------------------------------------
static inline char* carve(char*& p, size_t bytes) {
    char* r = p;
    p += (bytes + 255) & ~(size_t)255;
    return r;
}

extern "C" void kernel_launch(void* const* d_in, const int* in_sizes, int n_in,
                              void* d_out, int out_size, void* d_ws, size_t ws_size,
                              hipStream_t stream) {
    (void)in_sizes; (void)n_in; (void)out_size; (void)ws_size;
    const float* x_t   = (const float*)d_in[0];
    const float* H0    = (const float*)d_in[1];
    const float* C0    = (const float*)d_in[2];
    const float* m0    = (const float*)d_in[3];
    const float* R0    = (const float*)d_in[4];
    const float* A0    = (const float*)d_in[5];
    const float* Wprep = (const float*)d_in[6];
    const float* bprep = (const float*)d_in[7];
    const float* Wx    = (const float*)d_in[8];
    const float* Wh    = (const float*)d_in[9];
    const float* bl    = (const float*)d_in[10];
    const float* Wr    = (const float*)d_in[11];
    const float* br    = (const float*)d_in[12];
    const float* Ww    = (const float*)d_in[13];
    const float* bw    = (const float*)d_in[14];
    const float* Wo    = (const float*)d_in[15];
    const float* bo    = (const float*)d_in[16];

    float* out    = (float*)d_out;
    float* out_y  = out;                                      // [B,U]
    float* out_mt = out + (size_t)BATCH * UDIM;               // [B,N,M]
    float* out_Rt = out_mt + (size_t)BATCH * NMEM * MMEM;     // [B,M]
    float* out_wr = out_Rt + (size_t)BATCH * MMEM;            // [B,N]
    float* out_ww = out_wr + (size_t)BATCH * NMEM;            // [B,N]

    // workspace carve (~31 MB); all weight matrices stored TRANSPOSED [N,K]
    char* p = (char*)d_ws;
    unsigned short* Xbf    = (unsigned short*)carve(p, (size_t)BATCH * KBIGP * 2);
    unsigned short* WzT    = (unsigned short*)carve(p, (size_t)ZCOLS * KBIGP * 2);
    unsigned short* R0bf   = (unsigned short*)carve(p, (size_t)BATCH * MMEM * 2);
    unsigned short* WpT    = (unsigned short*)carve(p, (size_t)UDIM * MMEM * 2);
    float*          zbuf   = (float*)carve(p, (size_t)BATCH * ZCOLS * 4);
    unsigned short* Xo     = (unsigned short*)carve(p, (size_t)BATCH * XOCOLS * 2);
    unsigned short* WheadT = (unsigned short*)carve(p, (size_t)HEADP * UDIM * 2);
    float*          bhead  = (float*)carve(p, (size_t)HEADP * 4);
    float*          heads  = (float*)carve(p, (size_t)BATCH * HEADP * 4);
    unsigned short* WoT    = (unsigned short*)carve(p, (size_t)UDIM * XOCOLS * 2);

    dim3 blk(256);
    auto grid1 = [](long total) { return dim3((unsigned)((total + 255) / 256)); };

    // -- zero padded operands so the GEMM needs no bounds checks --
    hipMemsetAsync(Xbf,    0, (size_t)BATCH * KBIGP * 2, stream);
    hipMemsetAsync(WzT,    0, (size_t)ZCOLS * KBIGP * 2, stream);
    hipMemsetAsync(WheadT, 0, (size_t)HEADP * UDIM * 2, stream);
    hipMemsetAsync(bhead,  0, (size_t)HEADP * 4, stream);

    // -- pack bf16 operands (fuses all concats; weights transposed) --
    cvt_f32_bf16_kernel<<<grid1((long)BATCH * INDIM), blk, 0, stream>>>(x_t, INDIM, Xbf, KBIGP, BATCH, INDIM);
    cvt_f32_bf16_kernel<<<grid1((long)BATCH * UDIM),  blk, 0, stream>>>(H0, UDIM, Xbf + INDIM + UDIM, KBIGP, BATCH, UDIM);
    cvt_f32_bf16_kernel<<<grid1((long)BATCH * MMEM),  blk, 0, stream>>>(R0, MMEM, R0bf, MMEM, BATCH, MMEM);
    cvt_t_f32_bf16_kernel<<<grid1((long)MMEM * UDIM), blk, 0, stream>>>(Wprep, UDIM, MMEM, UDIM, WpT, MMEM);
    cvt_t_f32_bf16_kernel<<<grid1((long)(INDIM + UDIM) * ZCOLS), blk, 0, stream>>>(Wx, ZCOLS, INDIM + UDIM, ZCOLS, WzT, KBIGP);
    cvt_t_f32_bf16_kernel<<<grid1((long)UDIM * ZCOLS), blk, 0, stream>>>(Wh, ZCOLS, UDIM, ZCOLS, WzT + INDIM + UDIM, KBIGP);
    cvt_t_f32_bf16_kernel<<<grid1((long)UDIM * (MMEM + 6)), blk, 0, stream>>>(Wr, MMEM + 6, UDIM, MMEM + 6, WheadT, UDIM);
    cvt_t_f32_bf16_kernel<<<grid1((long)UDIM * (3 * MMEM + 6)), blk, 0, stream>>>(Ww, 3 * MMEM + 6, UDIM, 3 * MMEM + 6, WheadT + (size_t)(MMEM + 6) * UDIM, UDIM);
    cvt_t_f32_bf16_kernel<<<grid1((long)XOCOLS * UDIM), blk, 0, stream>>>(Wo, UDIM, XOCOLS, UDIM, WoT, XOCOLS);
    pack_bias_kernel<<<dim3(2), blk, 0, stream>>>(br, bw, bhead);

    // -- GEMM a: h_pre = R0 @ Wprep + bprep  -> bf16 into Xbf[:, 10:522] --
    gemm_bf16_wmma_kernel<<<dim3(UDIM / 64, BATCH / 128), blk, 0, stream>>>(
        R0bf, MMEM, WpT, MMEM, bprep, (void*)(Xbf + INDIM), KBIGP,
        MMEM, GEMM_OUT_BF16);

    // -- GEMM b: z = X @ [Wx;Wh] + bl   (the 8.7 GFLOP GEMM, K padded) --
    gemm_bf16_wmma_kernel<<<dim3(ZCOLS / 64, BATCH / 128), blk, 0, stream>>>(
        Xbf, KBIGP, WzT, KBIGP, bl, (void*)zbuf, ZCOLS,
        KBIGP, 0);

    // -- LSTM gates -> h (bf16) into Xo[:, 0:512] --
    lstm_kernel<<<grid1((long)BATCH * UDIM), blk, 0, stream>>>(zbuf, C0, Xo);

    // -- GEMM c: heads = h @ [Wr|Ww] + [br|bw]   (N padded to 320) --
    gemm_bf16_wmma_kernel<<<dim3(HEADP / 64, BATCH / 128), blk, 0, stream>>>(
        Xo, XOCOLS, WheadT, UDIM, bhead, (void*)heads, HEADP,
        UDIM, 0);

    // -- addressing + memory update (single pass over m0) --
    addressing_kernel<<<dim3(BATCH), blk, 0, stream>>>(
        heads, A0, m0, out_mt, out_Rt, out_wr, out_ww, Xo + UDIM);

    // -- GEMM d: y_t = clip([h|R_t] @ Wo + bo) --
    gemm_bf16_wmma_kernel<<<dim3(UDIM / 64, BATCH / 128), blk, 0, stream>>>(
        Xo, XOCOLS, WoT, XOCOLS, bo, (void*)out_y, UDIM,
        XOCOLS, GEMM_CLIP);
}
